// CrossAttention_67370857005263
// MI455X (gfx1250) — compile-verified, matrix operand order checked by reading, not run on hardware
//
#include <hip/hip_runtime.h>
#include <hip/hip_bf16.h>

// ---------------------------------------------------------------------------
// CrossAttention + WPR token pruning for MI455X (gfx1250), wave32 + WMMA bf16.
//   B=1, N=4096, QD=1152, H=16, D=72 (padded to 96), RETAIN=1638
// All GEMM operands padded + pre-transposed: branchless WMMA inner loop,
// 128-bit fragment loads, 32x32 register tile per wave (4 WMMAs / K-step).
// ---------------------------------------------------------------------------

#define N_TOK   4096
#define QDIM    1152
#define HEADS   16
#define DHEAD   72
#define DPAD    96           // DHEAD padded to multiple of 32
#define HDIM    1152         // HEADS * DHEAD
#define HDPAD   1536         // HEADS * DPAD
#define RETAINK 1638
#define PRUNEK  (N_TOK - RETAINK)   // 2458

typedef __attribute__((ext_vector_type(16))) __bf16 v16bf;
typedef __attribute__((ext_vector_type(8)))  __bf16 v8bf;
typedef __attribute__((ext_vector_type(8)))  float  v8f;

// ---------------------------------------------------------------------------
// casts / repacks / zero-fill
// ---------------------------------------------------------------------------
__global__ void k_cast_bf16(const float* __restrict__ in, __bf16* __restrict__ out, long n) {
    long i = (long)blockIdx.x * blockDim.x + threadIdx.x;
    long stride = (long)gridDim.x * blockDim.x;
    for (; i < n; i += stride) out[i] = (__bf16)in[i];
}

__global__ void k_zero_bf16(__bf16* __restrict__ out, long n) {
    long i = (long)blockIdx.x * blockDim.x + threadIdx.x;
    long stride = (long)gridDim.x * blockDim.x;
    for (; i < n; i += stride) out[i] = (__bf16)0.0f;
}

// W (K x N) f32 -> WT (N x K) bf16
__global__ void k_transpose_w(const float* __restrict__ W, __bf16* __restrict__ WT,
                              int K, int N) {
    long o = (long)blockIdx.x * blockDim.x + threadIdx.x;
    long total = (long)K * N;
    long stride = (long)gridDim.x * blockDim.x;
    for (; o < total; o += stride) {
        int n = (int)(o / K), k = (int)(o % K);
        WT[o] = (__bf16)W[(long)k * N + n];
    }
}

// Wo (1152 x 1152) f32 -> WoT (1152 x 1536) bf16, head-pad rows zeroed
__global__ void k_transpose_wo(const float* __restrict__ Wo, __bf16* __restrict__ WoT) {
    long o = (long)blockIdx.x * blockDim.x + threadIdx.x;
    long total = (long)QDIM * HDPAD;
    long stride = (long)gridDim.x * blockDim.x;
    for (; o < total; o += stride) {
        int n = (int)(o / HDPAD), kp = (int)(o % HDPAD);
        int h = kp / DPAD, d = kp - h * DPAD;
        WoT[o] = (d < DHEAD) ? (__bf16)Wo[(long)(h * DHEAD + d) * QDIM + n] : (__bf16)0.0f;
    }
}

// ---------------------------------------------------------------------------
// WMMA bf16 GEMM core. A: M x K row-major (lda). Bt: N x K row-major (ldbt),
// i.e. B[k][n] = Bt[n][k]. K mult of 32; M mult of 256; N mult of 32.
// 16-bit fragment (ISA 7.12.2): lane L holds row (L&15); K elements are two
// contiguous 8-element (16B) runs selected by khalf = L>>4.
// Wave tile: 32x32 (2x2 WMMAs per K-step) -> 16 flops/byte of operand fetch.
// ---------------------------------------------------------------------------
__device__ __forceinline__ v16bf load_frag(const __bf16* __restrict__ row, int kb, int khalf) {
    const v8bf lo = *(const v8bf*)(row + kb + khalf * 8);
    const v8bf hi = *(const v8bf*)(row + kb + 16 + khalf * 8);
    return __builtin_shufflevector(lo, hi, 0, 1, 2, 3, 4, 5, 6, 7,
                                   8, 9, 10, 11, 12, 13, 14, 15);
}

// store modes for bf16 output
//  0: plain row-major C[m*ldc + ncol]
//  1: head-padded   Cp[h][m][d]   (h = ncol/72, d = ncol%72, slab 4096x96)
//  2: head-padded-T Cp[h][d][m]   (slab 96x4096)
__device__ __forceinline__ void store_bf16(__bf16* __restrict__ C, int ldc, int mode,
                                           int m, int ncol, float v) {
    if (mode == 0) {
        C[(long)m * ldc + ncol] = (__bf16)v;
    } else if (mode == 1) {
        int h = ncol / DHEAD, d = ncol - h * DHEAD;
        C[((long)h * N_TOK + m) * DPAD + d] = (__bf16)v;
    } else {
        int h = ncol / DHEAD, d = ncol - h * DHEAD;
        C[((long)h * DPAD + d) * N_TOK + m] = (__bf16)v;
    }
}

// block = 256 thr = 8 waves; wave w: rows [(by*8+w)*32,+32), cols [bx*32,+32)
__global__ void k_gemm_bf16out(const __bf16* __restrict__ A, int lda,
                               const __bf16* __restrict__ Bt, int ldbt,
                               int K, __bf16* __restrict__ C, int ldc, int mode) {
    const int wave  = threadIdx.x >> 5;
    const int lane  = threadIdx.x & 31;
    const int ln15  = lane & 15;
    const int khalf = lane >> 4;
    const int mbase = (blockIdx.y * 8 + wave) * 32;
    const int nbase = blockIdx.x * 32;

    const __bf16* arow0 = A  + (long)(mbase + ln15) * lda;
    const __bf16* arow1 = A  + (long)(mbase + 16 + ln15) * lda;
    const __bf16* brow0 = Bt + (long)(nbase + ln15) * ldbt;
    const __bf16* brow1 = Bt + (long)(nbase + 16 + ln15) * ldbt;

    v8f acc00 = {}, acc01 = {}, acc10 = {}, acc11 = {};
    for (int kb = 0; kb < K; kb += 32) {
        const v16bf a0 = load_frag(arow0, kb, khalf);
        const v16bf a1 = load_frag(arow1, kb, khalf);
        const v16bf b0 = load_frag(brow0, kb, khalf);
        const v16bf b1 = load_frag(brow1, kb, khalf);
        acc00 = __builtin_amdgcn_wmma_f32_16x16x32_bf16(false, a0, false, b0,
                                                        (short)0, acc00, false, false);
        acc01 = __builtin_amdgcn_wmma_f32_16x16x32_bf16(false, a0, false, b1,
                                                        (short)0, acc01, false, false);
        acc10 = __builtin_amdgcn_wmma_f32_16x16x32_bf16(false, a1, false, b0,
                                                        (short)0, acc10, false, false);
        acc11 = __builtin_amdgcn_wmma_f32_16x16x32_bf16(false, a1, false, b1,
                                                        (short)0, acc11, false, false);
    }
#pragma unroll
    for (int r = 0; r < 8; ++r) {
        const int m0 = mbase + r + 8 * khalf;
        const int m1 = m0 + 16;
        store_bf16(C, ldc, mode, m0, nbase + ln15,      acc00[r]);
        store_bf16(C, ldc, mode, m0, nbase + 16 + ln15, acc01[r]);
        store_bf16(C, ldc, mode, m1, nbase + ln15,      acc10[r]);
        store_bf16(C, ldc, mode, m1, nbase + 16 + ln15, acc11[r]);
    }
}

__global__ void k_gemm_f32out(const __bf16* __restrict__ A, int lda,
                              const __bf16* __restrict__ Bt, int ldbt,
                              int K, float* __restrict__ C, int ldc,
                              const float* __restrict__ bias) {
    const int wave  = threadIdx.x >> 5;
    const int lane  = threadIdx.x & 31;
    const int ln15  = lane & 15;
    const int khalf = lane >> 4;
    const int mbase = (blockIdx.y * 8 + wave) * 32;
    const int nbase = blockIdx.x * 32;

    const __bf16* arow0 = A  + (long)(mbase + ln15) * lda;
    const __bf16* arow1 = A  + (long)(mbase + 16 + ln15) * lda;
    const __bf16* brow0 = Bt + (long)(nbase + ln15) * ldbt;
    const __bf16* brow1 = Bt + (long)(nbase + 16 + ln15) * ldbt;

    v8f acc00 = {}, acc01 = {}, acc10 = {}, acc11 = {};
    for (int kb = 0; kb < K; kb += 32) {
        const v16bf a0 = load_frag(arow0, kb, khalf);
        const v16bf a1 = load_frag(arow1, kb, khalf);
        const v16bf b0 = load_frag(brow0, kb, khalf);
        const v16bf b1 = load_frag(brow1, kb, khalf);
        acc00 = __builtin_amdgcn_wmma_f32_16x16x32_bf16(false, a0, false, b0,
                                                        (short)0, acc00, false, false);
        acc01 = __builtin_amdgcn_wmma_f32_16x16x32_bf16(false, a0, false, b1,
                                                        (short)0, acc01, false, false);
        acc10 = __builtin_amdgcn_wmma_f32_16x16x32_bf16(false, a1, false, b0,
                                                        (short)0, acc10, false, false);
        acc11 = __builtin_amdgcn_wmma_f32_16x16x32_bf16(false, a1, false, b1,
                                                        (short)0, acc11, false, false);
    }
    const float b0v = bias ? bias[nbase + ln15]      : 0.0f;
    const float b1v = bias ? bias[nbase + 16 + ln15] : 0.0f;
#pragma unroll
    for (int r = 0; r < 8; ++r) {
        const int m0 = mbase + r + 8 * khalf;
        const int m1 = m0 + 16;
        C[(long)m0 * ldc + nbase + ln15]      = acc00[r] + b0v;
        C[(long)m0 * ldc + nbase + 16 + ln15] = acc01[r] + b1v;
        C[(long)m1 * ldc + nbase + ln15]      = acc10[r] + b0v;
        C[(long)m1 * ldc + nbase + 16 + ln15] = acc11[r] + b1v;
    }
}

// ---------------------------------------------------------------------------
// Row softmax of scaled scores; writes bf16 probs P and accumulates head-mean
// into attn (first head stores, later heads add). One block (256 thr) per row.
// ---------------------------------------------------------------------------
__global__ void k_softmax_row(const float* __restrict__ S, __bf16* __restrict__ P,
                              float* __restrict__ attn, float scale, float wmean,
                              int first) {
    const int row = blockIdx.x;
    const float* s = S + (long)row * N_TOK;
    __shared__ float red[8];
    const int tid = threadIdx.x;

    float m = -3.4e38f;
    for (int j = tid; j < N_TOK; j += 256) m = fmaxf(m, s[j] * scale);
#pragma unroll
    for (int o = 16; o > 0; o >>= 1) m = fmaxf(m, __shfl_xor(m, o, 32));
    if ((tid & 31) == 0) red[tid >> 5] = m;
    __syncthreads();
    float mm = red[0];
#pragma unroll
    for (int w = 1; w < 8; ++w) mm = fmaxf(mm, red[w]);
    __syncthreads();

    float sum = 0.0f;
    for (int j = tid; j < N_TOK; j += 256) sum += __expf(s[j] * scale - mm);
#pragma unroll
    for (int o = 16; o > 0; o >>= 1) sum += __shfl_xor(sum, o, 32);
    if ((tid & 31) == 0) red[tid >> 5] = sum;
    __syncthreads();
    float tot = 0.0f;
#pragma unroll
    for (int w = 0; w < 8; ++w) tot += red[w];
    const float inv = 1.0f / tot;

    for (int j = tid; j < N_TOK; j += 256) {
        const float p = __expf(s[j] * scale - mm) * inv;
        P[(long)row * N_TOK + j] = (__bf16)p;
        float* a = attn + (long)row * N_TOK + j;
        if (first) *a = p * wmean; else *a += p * wmean;
    }
}

// ---------------------------------------------------------------------------
// Power iteration: out[j] = sum_i dist[i] * attn[i][j].
// dist (16 KB) staged into LDS with CDNA5 async global->LDS copies.
// ---------------------------------------------------------------------------
__global__ void k_init_dist(float* __restrict__ d) {
    int i = blockIdx.x * blockDim.x + threadIdx.x;
    if (i < N_TOK) d[i] = 1.0f / (float)N_TOK;
}

__global__ void k_gemv_row(const float* __restrict__ dist, const float* __restrict__ A,
                           float* __restrict__ out) {
    __shared__ float sdist[N_TOK];
#pragma unroll
    for (int it = 0; it < 4; ++it) {
        const int idx = (it * 256 + (int)threadIdx.x) * 4;   // 16B per async load
        unsigned lds_off = (unsigned)(unsigned long long)(&sdist[idx]);
        const float* g = dist + idx;
        asm volatile("global_load_async_to_lds_b128 %0, %1, off"
                     :: "v"(lds_off), "v"(g) : "memory");
    }
    asm volatile("s_wait_asynccnt 0" ::: "memory");
    __syncthreads();

    const int j = blockIdx.x * blockDim.x + threadIdx.x;
    if (j < N_TOK) {
        __builtin_prefetch(&A[j], 0, 0);   // global_prefetch_b8
        float acc = 0.0f;
        for (int i = 0; i < N_TOK; ++i) acc += sdist[i] * A[(long)i * N_TOK + j];
        out[j] = acc;
    }
}

// ---------------------------------------------------------------------------
// O(N^2) stable ranks (tie-break by lower index, matching jax.lax.top_k),
// permutation scatter, index-sorted member lists, argmax gather.
// ---------------------------------------------------------------------------
__global__ void k_rank(const float* __restrict__ v, int* __restrict__ drank,
                       int* __restrict__ arank) {
    const int i = blockIdx.x * blockDim.x + threadIdx.x;
    if (i >= N_TOK) return;
    const float vi = v[i];
    int dg = 0, ag = 0;
    for (int j = 0; j < N_TOK; ++j) {
        const float vj = v[j];
        dg += (vj > vi) || (vj == vi && j < i);
        ag += (vj < vi) || (vj == vi && j < i);
    }
    drank[i] = dg;
    arank[i] = ag;
}

__global__ void k_scatter_perm(const int* __restrict__ drank, const int* __restrict__ arank,
                               int* __restrict__ permd, int* __restrict__ perma) {
    const int i = blockIdx.x * blockDim.x + threadIdx.x;
    if (i >= N_TOK) return;
    permd[drank[i]] = i;
    perma[arank[i]] = i;
}

__global__ void k_sorted_members(const int* __restrict__ rank, int k, int* __restrict__ out) {
    const int i = blockIdx.x * blockDim.x + threadIdx.x;
    if (i >= N_TOK) return;
    if (rank[i] < k) {
        int pos = 0;
        for (int j = 0; j < i; ++j) pos += (rank[j] < k);
        out[pos] = i;
    }
}

__global__ void k_sim_inds(const float* __restrict__ attn,
                           const int* __restrict__ imp_sorted,
                           const int* __restrict__ prune_sorted,
                           int* __restrict__ out) {
    const int t = blockIdx.x * blockDim.x + threadIdx.x;
    if (t >= PRUNEK) return;
    const int j = prune_sorted[t];
    float best = -3.4e38f;
    int bestr = 0;
    for (int r = 0; r < RETAINK; ++r) {
        const float val = attn[(long)imp_sorted[r] * N_TOK + j];
        if (val > best) { best = val; bestr = r; }
    }
    out[t] = bestr;
}

__global__ void k_copy_f32(const float* __restrict__ in, float* __restrict__ out, int n) {
    const int i = blockIdx.x * blockDim.x + threadIdx.x;
    if (i < n) out[i] = in[i];
}
__global__ void k_copy_i32(const int* __restrict__ in, int* __restrict__ out, int n) {
    const int i = blockIdx.x * blockDim.x + threadIdx.x;
    if (i < n) out[i] = in[i];
}

// ---------------------------------------------------------------------------
// host orchestration
// ---------------------------------------------------------------------------
extern "C" void kernel_launch(void* const* d_in, const int* in_sizes, int n_in,
                              void* d_out, int out_size, void* d_ws, size_t ws_size,
                              hipStream_t stream) {
    const float* x  = (const float*)d_in[0];   // (4096, 1152)
    const float* Wq = (const float*)d_in[1];   // (1152, 1152)
    const float* Wk = (const float*)d_in[2];
    const float* Wv = (const float*)d_in[3];
    const float* Wo = (const float*)d_in[4];
    const float* bo = (const float*)d_in[5];   // (1152,)

    const float scale = 1.0f / sqrtf((float)DHEAD);

    // ---- workspace carve-up (~235 MB) ----
    char* w = (char*)d_ws;
    size_t off = 0;
    auto carve = [&](size_t bytes) -> void* {
        void* p = w + off;
        off = (off + bytes + 255) & ~(size_t)255;
        return p;
    };
    __bf16* xb   = (__bf16*)carve((size_t)N_TOK * QDIM * 2);
    __bf16* WqT  = (__bf16*)carve((size_t)QDIM * QDIM * 2);   // (N=1152, K=1152)
    __bf16* WkT  = (__bf16*)carve((size_t)QDIM * QDIM * 2);
    __bf16* WvT  = (__bf16*)carve((size_t)QDIM * QDIM * 2);
    __bf16* WoT  = (__bf16*)carve((size_t)QDIM * HDPAD * 2);  // (1152, 1536) padded
    __bf16* Qp   = (__bf16*)carve((size_t)HEADS * N_TOK * DPAD * 2);  // (16,4096,96)
    __bf16* Kp   = (__bf16*)carve((size_t)HEADS * N_TOK * DPAD * 2);
    __bf16* VpT  = (__bf16*)carve((size_t)HEADS * DPAD * N_TOK * 2);  // (16,96,4096)
    __bf16* Obp  = (__bf16*)carve((size_t)N_TOK * HDPAD * 2);         // (4096,1536)
    float*  Sb   = (float*) carve((size_t)N_TOK * N_TOK * 4);
    __bf16* Pb   = (__bf16*)carve((size_t)N_TOK * N_TOK * 2);
    float*  attn = (float*) carve((size_t)N_TOK * N_TOK * 4);
    float*  dA   = (float*) carve((size_t)N_TOK * 4);
    float*  dB   = (float*) carve((size_t)N_TOK * 4);
    int* drank   = (int*)carve((size_t)N_TOK * 4);
    int* arank   = (int*)carve((size_t)N_TOK * 4);
    int* permd   = (int*)carve((size_t)N_TOK * 4);
    int* perma   = (int*)carve((size_t)N_TOK * 4);
    int* impso   = (int*)carve((size_t)RETAINK * 4);
    int* pruso   = (int*)carve((size_t)PRUNEK * 4);

    // ---- output layout (concatenated flat, reference return order) ----
    float* out_o   = (float*)d_out;                       // 4096*1152
    float* out_imp = out_o + (size_t)N_TOK * QDIM;        // 4096
    int*   out_ii  = (int*)(out_imp + N_TOK);             // 1638
    int*   out_pi  = out_ii + RETAINK;                    // 2458
    int*   out_si  = out_pi + PRUNEK;                     // 2458

    // 1) input cast + weight transpose-repacks + zero padded buffers
    k_cast_bf16   <<<1024, 256, 0, stream>>>(x, xb, (long)N_TOK * QDIM);
    k_transpose_w <<<1024, 256, 0, stream>>>(Wq, WqT, QDIM, QDIM);
    k_transpose_w <<<1024, 256, 0, stream>>>(Wk, WkT, QDIM, QDIM);
    k_transpose_w <<<1024, 256, 0, stream>>>(Wv, WvT, QDIM, QDIM);
    k_transpose_wo<<<1024, 256, 0, stream>>>(Wo, WoT);
    k_zero_bf16   <<<1024, 256, 0, stream>>>(Qp,  (long)HEADS * N_TOK * DPAD);
    k_zero_bf16   <<<1024, 256, 0, stream>>>(Kp,  (long)HEADS * N_TOK * DPAD);
    k_zero_bf16   <<<1024, 256, 0, stream>>>(VpT, (long)HEADS * DPAD * N_TOK);

    // 2) Q/K/V projections: (4096x1152)@(1152x1152); stores into padded layouts
    {
        dim3 g(QDIM / 32, N_TOK / 256);
        k_gemm_bf16out<<<g, 256, 0, stream>>>(xb, QDIM, WqT, QDIM, QDIM, Qp,  0, 1);
        k_gemm_bf16out<<<g, 256, 0, stream>>>(xb, QDIM, WkT, QDIM, QDIM, Kp,  0, 1);
        k_gemm_bf16out<<<g, 256, 0, stream>>>(xb, QDIM, WvT, QDIM, QDIM, VpT, 0, 2);
    }

    // 3) per-head streaming attention
    for (int h = 0; h < HEADS; ++h) {
        const __bf16* Qh  = Qp  + (size_t)h * N_TOK * DPAD;   // A: 4096 x 96
        const __bf16* Kh  = Kp  + (size_t)h * N_TOK * DPAD;   // Bt: 4096 x 96
        const __bf16* VhT = VpT + (size_t)h * DPAD * N_TOK;   // Bt: 96 x 4096
        // sim = Q_h @ K_h^T   (4096 x 4096, K=96 padded)
        {
            dim3 g(N_TOK / 32, N_TOK / 256);
            k_gemm_f32out<<<g, 256, 0, stream>>>(Qh, DPAD, Kh, DPAD, DPAD,
                                                 Sb, N_TOK, nullptr);
        }
        // softmax rows -> P (bf16); attn += p/16
        k_softmax_row<<<N_TOK, 256, 0, stream>>>(Sb, Pb, attn, scale,
                                                 1.0f / (float)HEADS, h == 0);
        // out_h = P @ V_h  (4096 x 96, K=4096) -> cols [h*96, h*96+96) of Obp
        {
            dim3 g(DPAD / 32, N_TOK / 256);
            k_gemm_bf16out<<<g, 256, 0, stream>>>(Pb, N_TOK, VhT, N_TOK, N_TOK,
                                                  Obp + h * DPAD, HDPAD, 0);
        }
    }

    // 4) power iteration: dist @ attn, 5 times (async-LDS staged GEMV)
    k_init_dist<<<16, 256, 0, stream>>>(dA);
    k_gemv_row<<<16, 256, 0, stream>>>(dA, attn, dB);
    k_gemv_row<<<16, 256, 0, stream>>>(dB, attn, dA);
    k_gemv_row<<<16, 256, 0, stream>>>(dA, attn, dB);
    k_gemv_row<<<16, 256, 0, stream>>>(dB, attn, dA);
    k_gemv_row<<<16, 256, 0, stream>>>(dA, attn, dB);   // final importance in dB
    k_copy_f32<<<16, 256, 0, stream>>>(dB, out_imp, N_TOK);

    // 5) ranks / permutations / index-sorted member lists
    k_rank<<<16, 256, 0, stream>>>(dB, drank, arank);
    k_scatter_perm<<<16, 256, 0, stream>>>(drank, arank, permd, perma);
    k_copy_i32<<<(RETAINK + 255) / 256, 256, 0, stream>>>(permd, out_ii, RETAINK);
    k_copy_i32<<<(PRUNEK + 255) / 256, 256, 0, stream>>>(perma, out_pi, PRUNEK);
    k_sorted_members<<<16, 256, 0, stream>>>(drank, RETAINK, impso);
    k_sorted_members<<<16, 256, 0, stream>>>(arank, PRUNEK, pruso);

    // 6) sim_inds: argmax over retained rows at pruned columns
    k_sim_inds<<<(PRUNEK + 255) / 256, 256, 0, stream>>>(attn, impso, pruso, out_si);

    // 7) output projection: Obp(4096x1536) @ WoT^T + bo -> f32 out (pad rows of
    //    WoT are zero so padded columns of Obp contribute nothing)
    {
        dim3 g(QDIM / 32, N_TOK / 256);
        k_gemm_f32out<<<g, 256, 0, stream>>>(Obp, HDPAD, WoT, HDPAD, HDPAD,
                                             out_o, QDIM, bo);
    }
    (void)in_sizes; (void)n_in; (void)out_size; (void)ws_size;
}
